// Transformer_56444460204165
// MI455X (gfx1250) — matrix operand order repaired and checked by reading, not compiled, not run on hardware
//
#include <hip/hip_runtime.h>
#include <math.h>

// ---------------- problem constants ----------------
#define NN      50000
#define KK      8
#define EE      200000
#define INF_    128
#define KEYF    4
#define FF      132          // attention feature dim
#define FPAD    160          // F padded to 5*32 (wmma k-steps)
#define NPAD    144          // F padded to 9*16 (wmma n-tiles)
#define HH      4
#define DD      33           // F / H
#define HID     256
#define D0      260          // 2*IN + KEY
#define D0PAD   288          // 9*32
#define MT_QKV  25000        // (N*K)/16 ; 2 nodes per M-tile
#define ET      12500        // E/16

typedef __attribute__((ext_vector_type(16))) unsigned short v16u;
typedef __attribute__((ext_vector_type(16))) __bf16         v16bf;
typedef __attribute__((ext_vector_type(8)))  float          v8f;

static __device__ __forceinline__ unsigned short f2bf(float f) {
  union { float f; unsigned u; } x; x.f = f;
  unsigned r = x.u + 0x7FFFu + ((x.u >> 16) & 1u);   // round-to-nearest-even
  return (unsigned short)(r >> 16);
}
static __device__ __forceinline__ v16bf ldfrag(const unsigned short* p) {
  return __builtin_bit_cast(v16bf, *(const v16u*)p);
}

// A-fragment (16x32 bf16) k-index for lane `lane`, element t (0..15); row m = lane&15.
static __device__ __forceinline__ int a_kidx(int lane, int t) {
  int hi = lane >> 4;
  return hi * 8 + ((t >> 1) & 3) * 2 + (t >> 3) * 16 + (t & 1);
}

// ---------------- weight packing: B-fragment layout [nt][kt][lane32][16] ----------------
__global__ void pack_w(const float* __restrict__ W, int Kdim, int Ndim,
                       int KT, int NT, unsigned short* __restrict__ dst) {
  int idx = blockIdx.x * blockDim.x + threadIdx.x;
  int total = NT * KT * 512;
  if (idx >= total) return;
  int t    = idx & 15;
  int lane = (idx >> 4) & 31;
  int kt   = (idx >> 9) % KT;
  int nt   = (idx >> 9) / KT;
  int n = nt * 16 + (lane & 15);
  int k = kt * 32 + (lane >> 4) * 8 + (t >> 3) * 16 + (t & 7);
  float v = (k < Kdim && n < Ndim) ? W[(size_t)k * Ndim + n] : 0.f;
  dst[idx] = f2bf(v);
}

// ---------------- features: gather in-edges, key stats, emit A-packed x ----------------
__global__ void features_kernel(const float* __restrict__ edges,
                                const float* __restrict__ word_mean,
                                const float* __restrict__ word_std,
                                const float* __restrict__ dists,
                                const int*   __restrict__ in_idx,
                                const int*   __restrict__ in_mask,
                                unsigned short* __restrict__ xA) {
  int mt = blockIdx.x, t = threadIdx.x;        // 128 threads; 2 nodes = 16 rows
  __shared__ float xr[16][FPAD];
  __shared__ float post[16], pmS[2], psS[2];
  __shared__ int   mk[16];
  if (t < 16) mk[t] = in_mask[mt * 16 + t];
  for (int r = 0; r < 16; ++r) {
    int e = in_idx[mt * 16 + r];
    xr[r][t] = edges[(size_t)e * INF_ + t];
  }
  __syncthreads();
  if (t < 16) post[t] = xr[t][INF_ - 1] * word_std[INF_ - 1] + word_mean[INF_ - 1];
  __syncthreads();
  if (t < 2) {
    float cnt = 0.f, s = 0.f;
    for (int j = 0; j < 8; ++j) { float m = (float)mk[t * 8 + j]; cnt += m; s += m * post[t * 8 + j]; }
    float pm = s / fmaxf(cnt, 1.f);
    float sv = 0.f;
    for (int j = 0; j < 8; ++j) { float m = (float)mk[t * 8 + j]; float d = post[t * 8 + j] - pm; sv += m * d * d; }
    pmS[t] = pm;
    psS[t] = sqrtf(sv / fmaxf(cnt - 1.f, 1.f));
  }
  __syncthreads();
  if (t < 16) {
    int node = t >> 3;
    xr[t][INF_ + 0] = post[t];
    xr[t][INF_ + 1] = pmS[node];
    xr[t][INF_ + 2] = psS[node];
    xr[t][INF_ + 3] = dists[mt * 16 + t];
    for (int c = FF; c < FPAD; ++c) xr[t][c] = 0.f;
  }
  __syncthreads();
  for (int i = t; i < 5 * 512; i += 128) {
    int e16 = i & 15, lane = (i >> 4) & 31, kt = i >> 9;
    int m = lane & 15;
    int k = kt * 32 + a_kidx(lane, e16);
    xA[((size_t)(mt * 5 + kt) * 32 + lane) * 16 + e16] = f2bf(xr[m][k]);
  }
}

// ---------------- QKV (WMMA) + attention + masked pooling ----------------
__global__ void qkv_attn_kernel(const unsigned short* __restrict__ xA,
                                const unsigned short* __restrict__ wqP,
                                const unsigned short* __restrict__ wkP,
                                const unsigned short* __restrict__ wvP,
                                const int* __restrict__ in_mask,
                                unsigned short* __restrict__ pooledB) {
  int mt = blockIdx.x, lane = threadIdx.x;     // 32 threads = 1 wave, 2 nodes
  __shared__ float sQKV[3][16][NPAD];
  __shared__ float sO[16][FF];
  __shared__ int   mk[16];
  if (lane < 16) mk[lane] = in_mask[mt * 16 + lane];

  v16bf a[5];
#pragma unroll
  for (int kt = 0; kt < 5; ++kt)
    a[kt] = ldfrag(xA + ((size_t)(mt * 5 + kt) * 32 + lane) * 16);

  int hi = lane >> 4, n = lane & 15;
#pragma unroll
  for (int w = 0; w < 3; ++w) {
    const unsigned short* wp = (w == 0) ? wqP : (w == 1) ? wkP : wvP;
    for (int nt = 0; nt < 9; ++nt) {
      v8f acc = {};
#pragma unroll
      for (int kt = 0; kt < 5; ++kt) {
        v16bf b = ldfrag(wp + ((size_t)(nt * 5 + kt) * 32 + lane) * 16);
        acc = __builtin_amdgcn_wmma_f32_16x16x32_bf16(false, a[kt], false, b,
                                                      (short)0, acc, false, false);
      }
#pragma unroll
      for (int g = 0; g < 8; ++g)
        sQKV[w][g + 8 * hi][nt * 16 + n] = acc[g];
    }
  }
  __syncthreads();

  // attention: lane -> (head h, query row i); 2 nodes sequentially
  int h = lane >> 3, i = lane & 7;
  const float scale = 0.17407765595569785f;    // 1/sqrt(33)
  for (int node = 0; node < 2; ++node) {
    int rb = node * 8;
    float sc[8], mx = -3.0e38f;
    for (int j = 0; j < 8; ++j) {
      float s = 0.f;
      for (int c = 0; c < DD; ++c)
        s += sQKV[0][rb + i][h * DD + c] * sQKV[1][rb + j][h * DD + c];
      s *= scale;
      if (mk[rb + j] == 0) s = -1.0e9f;
      sc[j] = s; mx = fmaxf(mx, s);
    }
    float den = 0.f;
    for (int j = 0; j < 8; ++j) { sc[j] = __expf(sc[j] - mx); den += sc[j]; }
    float inv = 1.f / den;
    for (int c = 0; c < DD; ++c) {
      float o = 0.f;
      for (int j = 0; j < 8; ++j) o += sc[j] * sQKV[2][rb + j][h * DD + c];
      sO[rb + i][h * DD + c] = o * inv;
    }
  }
  __syncthreads();

  // masked mean-pool over K; zero when no in-edges
  for (int node = 0; node < 2; ++node) {
    float cnt = 0.f;
    for (int j = 0; j < 8; ++j) cnt += (float)mk[node * 8 + j];
    float invd = (cnt > 0.f) ? 1.f / fmaxf(cnt, 1.f) : 0.f;
    for (int f = lane; f < FF; f += 32) {
      float s = 0.f;
      for (int j = 0; j < 8; ++j) s += (float)mk[node * 8 + j] * sO[node * 8 + j][f];
      pooledB[((size_t)(2 * mt + node)) * FF + f] = f2bf(s * invd);
    }
  }
}

// ---------------- build MLP input [E, 288] in A-packed bf16 ----------------
__global__ void build_h_kernel(const float* __restrict__ edges,
                               const unsigned short* __restrict__ pooledB,
                               const int* __restrict__ esrc,
                               unsigned short* __restrict__ hA) {
  int mt = blockIdx.x, t = threadIdx.x;        // 128 threads, 16 edges
  __shared__ unsigned short hr[16][D0PAD];
  __shared__ int srcs[16];
  if (t < 16) srcs[t] = esrc[mt * 16 + t];
  __syncthreads();
  for (int r = 0; r < 16; ++r)
    hr[r][t] = f2bf(edges[((size_t)mt * 16 + r) * INF_ + t]);     // cols 0..127
  for (int i = t; i < 16 * 160; i += 128) {                        // cols 128..287
    int r = i / 160, c = i % 160;
    hr[r][INF_ + c] = (c < FF) ? pooledB[(size_t)srcs[r] * FF + c] : (unsigned short)0;
  }
  __syncthreads();
  for (int i = t; i < 9 * 512; i += 128) {
    int e16 = i & 15, lane = (i >> 4) & 31, kt = i >> 9;
    int m = lane & 15;
    int k = kt * 32 + a_kidx(lane, e16);
    hA[((size_t)(mt * 9 + kt) * 32 + lane) * 16 + e16] = hr[m][k];
  }
}

// ---------------- generic WMMA MLP layer: [E, KT*32] x [KT*32, 256] ----------------
template<int KT, bool RELU, bool PACK_OUT>
__global__ void mlp_gemm_kernel(const unsigned short* __restrict__ Apack,
                                const unsigned short* __restrict__ Wpack,
                                const float* __restrict__ bias,
                                unsigned short* __restrict__ outPack,
                                float* __restrict__ outRow) {
  int mt = blockIdx.x, lane = threadIdx.x;     // 32 threads = 1 wave
  __shared__ unsigned short sOut[16][HID];
  v16bf a[KT];
#pragma unroll
  for (int kt = 0; kt < KT; ++kt)
    a[kt] = ldfrag(Apack + ((size_t)(mt * KT + kt) * 32 + lane) * 16);

  int hi = lane >> 4, n = lane & 15;
  for (int nt = 0; nt < 16; ++nt) {
    v8f acc = {};
#pragma unroll
    for (int kt = 0; kt < KT; ++kt) {
      v16bf b = ldfrag(Wpack + ((size_t)(nt * KT + kt) * 32 + lane) * 16);
      acc = __builtin_amdgcn_wmma_f32_16x16x32_bf16(false, a[kt], false, b,
                                                    (short)0, acc, false, false);
    }
    float bv = bias[nt * 16 + n];
#pragma unroll
    for (int g = 0; g < 8; ++g) {
      float v = acc[g] + bv;
      if (RELU) v = v > 0.f ? v : 0.f;
      int m = g + 8 * hi;
      if (PACK_OUT) sOut[m][nt * 16 + n] = f2bf(v);
      else          outRow[((size_t)(mt * 16 + m)) * HID + nt * 16 + n] = v;
    }
  }
  if (PACK_OUT) {
    __syncthreads();
    int m = lane & 15;
#pragma unroll
    for (int kt = 0; kt < 8; ++kt) {
      v16u tmp;
#pragma unroll
      for (int t = 0; t < 16; ++t)
        tmp[t] = sOut[m][kt * 32 + a_kidx(lane, t)];
      *(v16u*)(outPack + ((size_t)(mt * 8 + kt) * 32 + lane) * 16) = tmp;
    }
  }
}

// ---------------- workspace layout (bytes) ----------------
#define OFF_WQ   ((size_t)0)
#define SZ_QKVW  ((size_t)5 * 9 * 512 * 2)                 // 46080
#define OFF_WK   (OFF_WQ + SZ_QKVW)
#define OFF_WV   (OFF_WK + SZ_QKVW)
#define OFF_W0   (OFF_WV + SZ_QKVW)
#define SZ_W0    ((size_t)9 * 16 * 512 * 2)                // 147456
#define OFF_W1   (OFF_W0 + SZ_W0)
#define SZ_W256  ((size_t)8 * 16 * 512 * 2)                // 131072
#define OFF_WO   (OFF_W1 + SZ_W256)
#define OFF_POOL (OFF_WO + SZ_W256)
#define SZ_POOL  ((size_t)NN * FF * 2)                     // 13.2 MB
#define OFF_XA   (OFF_POOL + SZ_POOL)                      // xA, reused as h1
#define SZ_XA    ((size_t)MT_QKV * 5 * 512 * 2)            // 128 MB
#define OFF_HA   (OFF_XA + SZ_XA)                          // hA, reused as h2
#define SZ_HA    ((size_t)ET * 9 * 512 * 2)                // 115.2 MB

extern "C" void kernel_launch(void* const* d_in, const int* in_sizes, int n_in,
                              void* d_out, int out_size, void* d_ws, size_t ws_size,
                              hipStream_t stream) {
  const float* edges     = (const float*)d_in[0];
  const float* word_mean = (const float*)d_in[1];
  const float* word_std  = (const float*)d_in[2];
  const float* dists     = (const float*)d_in[3];
  const float* Wq        = (const float*)d_in[4];
  const float* Wk        = (const float*)d_in[5];
  const float* Wv        = (const float*)d_in[6];
  const float* W0        = (const float*)d_in[7];
  const float* b0        = (const float*)d_in[8];
  const float* W1        = (const float*)d_in[9];
  const float* b1        = (const float*)d_in[10];
  const float* Wout      = (const float*)d_in[11];
  const float* bout      = (const float*)d_in[12];
  const int*   in_idx    = (const int*)d_in[13];
  const int*   in_mask   = (const int*)d_in[14];
  const int*   esrc      = (const int*)d_in[15];

  char* ws = (char*)d_ws;
  unsigned short* wqP  = (unsigned short*)(ws + OFF_WQ);
  unsigned short* wkP  = (unsigned short*)(ws + OFF_WK);
  unsigned short* wvP  = (unsigned short*)(ws + OFF_WV);
  unsigned short* w0P  = (unsigned short*)(ws + OFF_W0);
  unsigned short* w1P  = (unsigned short*)(ws + OFF_W1);
  unsigned short* woP  = (unsigned short*)(ws + OFF_WO);
  unsigned short* pool = (unsigned short*)(ws + OFF_POOL);
  unsigned short* xA   = (unsigned short*)(ws + OFF_XA);   // also h1
  unsigned short* hA   = (unsigned short*)(ws + OFF_HA);   // also h2

  // 1) pack weights into WMMA B-fragment layout (bf16)
  pack_w<<<(9 * 5 * 512 + 255) / 256, 256, 0, stream>>>(Wq, FF, FF, 5, 9, wqP);
  pack_w<<<(9 * 5 * 512 + 255) / 256, 256, 0, stream>>>(Wk, FF, FF, 5, 9, wkP);
  pack_w<<<(9 * 5 * 512 + 255) / 256, 256, 0, stream>>>(Wv, FF, FF, 5, 9, wvP);
  pack_w<<<(16 * 9 * 512 + 255) / 256, 256, 0, stream>>>(W0, D0, HID, 9, 16, w0P);
  pack_w<<<(16 * 8 * 512 + 255) / 256, 256, 0, stream>>>(W1, HID, HID, 8, 16, w1P);
  pack_w<<<(16 * 8 * 512 + 255) / 256, 256, 0, stream>>>(Wout, HID, HID, 8, 16, woP);

  // 2) gather + key features -> A-packed x
  features_kernel<<<MT_QKV, 128, 0, stream>>>(edges, word_mean, word_std, dists,
                                              in_idx, in_mask, xA);
  // 3) QKV (WMMA) + attention + pooling -> pooled bf16 [N, 132]
  qkv_attn_kernel<<<MT_QKV, 32, 0, stream>>>(xA, wqP, wkP, wvP, in_mask, pool);
  // 4) per-edge concat(edge, pooled[src]) -> A-packed [E, 288]
  build_h_kernel<<<ET, 128, 0, stream>>>(edges, pool, esrc, hA);
  // 5) MLP: layer0 (relu) -> layer1 (relu) -> out
  mlp_gemm_kernel<9, true,  true ><<<ET, 32, 0, stream>>>(hA, w0P, b0, xA, nullptr);
  mlp_gemm_kernel<8, true,  true ><<<ET, 32, 0, stream>>>(xA, w1P, b1, hA, nullptr);
  mlp_gemm_kernel<8, false, false><<<ET, 32, 0, stream>>>(hA, woP, bout, nullptr,
                                                          (float*)d_out);
}
